// EmformerEncoder_28733331210927
// MI455X (gfx1250) — matrix operand order repaired and checked by reading, not compiled, not run on hardware
//
#include <hip/hip_runtime.h>

typedef __attribute__((ext_vector_type(8)))  float  v8f;
typedef __attribute__((ext_vector_type(16))) __bf16 v16bf;
typedef __attribute__((ext_vector_type(8)))  __bf16 bf16x8;
typedef unsigned short u16;

#define S_LEN 2048
#define D_DIM 512
#define H_NUM 8
#define DK_DIM 64
#define FFN_DIM 2048
#define L_NUM 4
#define B_NUM 2
#define M_ROWS (B_NUM * S_LEN)

// ---------------- helpers ----------------

__device__ __forceinline__ u16 f2bf(float f) {
    unsigned int u = __float_as_uint(f);
    u += 0x7FFFu + ((u >> 16) & 1u);      // round-to-nearest-even
    return (u16)(u >> 16);
}

// A-operand fragment (16x32, 16-bit): lane half h holds K = h*16..h*16+15 (16 contiguous)
__device__ __forceinline__ v16bf ldA(const u16* p) {
    union { v16bf v; bf16x8 h[2]; } ua;
    ua.h[0] = *(const bf16x8*)(p);
    ua.h[1] = *(const bf16x8*)(p + 8);
    return ua.v;
}
// B-operand fragment (32x16, 16-bit): lane half h holds K = h*16..h*16+15 (16 contiguous)
__device__ __forceinline__ v16bf ldB(const u16* p) {
    return *(const v16bf*)(p);
}

__device__ __forceinline__ v8f wmma_bf16(v16bf a, v16bf b, v8f c) {
    return __builtin_amdgcn_wmma_f32_16x16x32_bf16(false, a, false, b, (short)0, c,
                                                   false, false);
}

// ---------------- weight transpose + f32->bf16 cast ----------------
// W[K,N] f32 (row major) -> Wt[N,K] bf16.  grid(N/32, K/32), block 256 (32x8)
__global__ void __launch_bounds__(256) transpose_cast_kernel(
    const float* __restrict__ W, u16* __restrict__ Wt, int K, int N)
{
    __shared__ float tile[32][33];
    const int n0 = blockIdx.x * 32, k0 = blockIdx.y * 32;
    const int tx = threadIdx.x & 31, ty = threadIdx.x >> 5;
#pragma unroll
    for (int j = 0; j < 4; ++j)
        tile[ty + j * 8][tx] = W[(size_t)(k0 + ty + j * 8) * N + (n0 + tx)];
    __syncthreads();
#pragma unroll
    for (int j = 0; j < 4; ++j)
        Wt[(size_t)(n0 + ty + j * 8) * K + (k0 + tx)] = f2bf(tile[tx][ty + j * 8]);
}

// ---------------- layernorm over D=512, optional f32/bf16 outputs ----------------
__global__ void __launch_bounds__(256) ln512_kernel(
    const float* __restrict__ x, const float* __restrict__ g, const float* __restrict__ bta,
    float* __restrict__ outF, u16* __restrict__ outB)
{
    const int row = blockIdx.x, t = threadIdx.x;
    const float* xr = x + (size_t)row * D_DIM;
    float v0 = xr[t], v1 = xr[t + 256];
    __shared__ float red[256];
    red[t] = v0 + v1;
    __syncthreads();
    for (int o = 128; o > 0; o >>= 1) { if (t < o) red[t] += red[t + o]; __syncthreads(); }
    const float mu = red[0] * (1.0f / 512.0f);
    __syncthreads();
    const float d0 = v0 - mu, d1 = v1 - mu;
    red[t] = d0 * d0 + d1 * d1;
    __syncthreads();
    for (int o = 128; o > 0; o >>= 1) { if (t < o) red[t] += red[t + o]; __syncthreads(); }
    const float rs = rsqrtf(red[0] * (1.0f / 512.0f) + 1e-3f);
    const float y0 = d0 * rs * g[t] + bta[t];
    const float y1 = d1 * rs * g[t + 256] + bta[t + 256];
    const size_t base = (size_t)row * D_DIM;
    if (outF) { outF[base + t] = y0; outF[base + t + 256] = y1; }
    if (outB) { outB[base + t] = f2bf(y0); outB[base + t + 256] = f2bf(y1); }
}

// ---------------- WMMA GEMM: C[M,N] = A[M,K](bf16) x Wt[N,K](bf16)^T + bias ----------------
// grid(N/128, M/128), block 128 (4 waves, 2x2); wave -> 64x64 tile (4x4 of 16x16, K-step 32)
// 16 b128 loads feed 16 WMMAs per K-step: 1:1 load:wmma, 4x fragment reuse.
__global__ void __launch_bounds__(128) gemm_bf16_wmma(
    const u16* __restrict__ A, const u16* __restrict__ Wt,
    const float* __restrict__ bias, const float* __restrict__ resid,
    float* __restrict__ outF, u16* __restrict__ outB,
    int M, int N, int K, int doRelu)
{
    const int lane = threadIdx.x & 31;
    const int wv   = threadIdx.x >> 5;               // 0..3
    const int m0   = blockIdx.y * 128 + (wv >> 1) * 64;
    const int n0   = blockIdx.x * 128 + (wv & 1) * 64;
    const int half = lane >> 4;
    const int mn   = lane & 15;

    v8f acc[4][4];
    {
        v8f z = {};
#pragma unroll
        for (int tm = 0; tm < 4; ++tm)
#pragma unroll
            for (int tn = 0; tn < 4; ++tn) acc[tm][tn] = z;
    }

    const u16* pa = A  + (size_t)(m0 + mn) * K + half * 16;
    const u16* pb = Wt + (size_t)(n0 + mn) * K + half * 16;
    const size_t rowK16 = (size_t)16 * K;

    for (int k0 = 0; k0 < K; k0 += 32) {
        v16bf af[4], bf_[4];
#pragma unroll
        for (int t = 0; t < 4; ++t) {
            af[t]  = ldA(pa + t * rowK16 + k0);
            bf_[t] = ldB(pb + t * rowK16 + k0);
        }
#pragma unroll
        for (int tm = 0; tm < 4; ++tm)
#pragma unroll
            for (int tn = 0; tn < 4; ++tn)
                acc[tm][tn] = wmma_bf16(af[tm], bf_[tn], acc[tm][tn]);
    }

#pragma unroll
    for (int tm = 0; tm < 4; ++tm) {
#pragma unroll
        for (int tn = 0; tn < 4; ++tn) {
            const int col = n0 + tn * 16 + mn;
            const float bb = bias[col];
#pragma unroll
            for (int r = 0; r < 8; ++r) {
                const int row = m0 + tm * 16 + half * 8 + r;
                const size_t gi = (size_t)row * N + col;
                float v = acc[tm][tn][r] + bb;
                if (doRelu) v = fmaxf(v, 0.0f);
                if (resid)  v += resid[gi];
                if (outF)   outF[gi] = v;
                if (outB)   outB[gi] = f2bf(v);
            }
        }
    }
}

// ---------------- banded attention for one (b, h, chunk), WT = window/16 ----------------
// block = 128 threads (4 waves); wave w handles rows [c*64 + w*16, +16)
template <int WT>
__global__ void __launch_bounds__(128) attn_wmma_kernel(
    const u16* __restrict__ Qb, const u16* __restrict__ Kb, const u16* __restrict__ Vb,
    const float* __restrict__ xin, float* __restrict__ res, int cBase)
{
    constexpr int W   = WT * 16;
    constexpr int STR = W + 16;                       // row stride (u16), 32B multiple
    __shared__ alignas(32) u16 sVt[64 * STR];         // V^T window: [dk][pos]
    __shared__ alignas(32) u16 sP [64 * STR];         // probs:      [row][pos]

    const int c    = cBase + (int)blockIdx.x;
    const int h    = blockIdx.y;
    const int b    = blockIdx.z;
    const int tid  = threadIdx.x;
    const int lane = tid & 31, wv = tid >> 5;
    const int half = lane >> 4, mn = lane & 15;
    int c0 = c * 64 - 128; if (c0 < 0) c0 = 0;

    // ---- stage V window transposed into LDS (coalesced 16B global reads) ----
#pragma unroll
    for (int it = 0; it < WT; ++it) {
        const int pos = it * 16 + (tid >> 3);
        const int dk0 = (tid & 7) * 8;
        const u16* src = Vb + (size_t)(b * S_LEN + c0 + pos) * D_DIM + h * DK_DIM + dk0;
        union { uint4 q; u16 s[8]; } u;
        u.q = *(const uint4*)src;
#pragma unroll
        for (int e = 0; e < 8; ++e) sVt[(dk0 + e) * STR + pos] = u.s[e];
    }
    __syncthreads();

    // ---- Q fragments (A operand), rows r0w..r0w+15, dk split 0..31 / 32..63 ----
    const int r0w = c * 64 + wv * 16;
    const size_t qrow = (size_t)(b * S_LEN + r0w + mn) * D_DIM + h * DK_DIM;
    const v16bf qf0 = ldA(Qb + qrow + half * 16);
    const v16bf qf1 = ldA(Qb + qrow + 32 + half * 16);

    // ---- scores = Q . K^T over the window (K rows are contiguous-dk B operands) ----
    v8f sc[WT];
    {
        v8f z = {};
#pragma unroll
        for (int t = 0; t < WT; ++t) sc[t] = z;
    }
#pragma unroll
    for (int t = 0; t < WT; ++t) {
        const size_t krow = (size_t)(b * S_LEN + c0 + t * 16 + mn) * D_DIM + h * DK_DIM + half * 16;
        const v16bf k0f = ldB(Kb + krow);
        const v16bf k1f = ldB(Kb + krow + 32);
        sc[t] = wmma_bf16(qf0, k0f, sc[t]);
        sc[t] = wmma_bf16(qf1, k1f, sc[t]);
    }

    // ---- softmax across the window (rows live in 16-lane halves of the C layout) ----
    float inv_sum[8];
#pragma unroll
    for (int r = 0; r < 8; ++r) {
        float mx = -3.0e38f;
#pragma unroll
        for (int t = 0; t < WT; ++t) mx = fmaxf(mx, sc[t][r]);
#pragma unroll
        for (int m = 1; m <= 8; m <<= 1) mx = fmaxf(mx, __shfl_xor(mx, m, 32));
        float sum = 0.0f;
#pragma unroll
        for (int t = 0; t < WT; ++t) {
            const float p = __expf((sc[t][r] - mx) * 0.125f);   // 1/sqrt(64) scale
            sc[t][r] = p;
            sum += p;
        }
#pragma unroll
        for (int m = 1; m <= 8; m <<= 1) sum += __shfl_xor(sum, m, 32);
        inv_sum[r] = 1.0f / sum;
    }

    // ---- C-layout -> A-layout via LDS (unnormalized probs as bf16) ----
    u16* pbase = sP + wv * 16 * STR;
#pragma unroll
    for (int t = 0; t < WT; ++t) {
#pragma unroll
        for (int r = 0; r < 8; ++r)
            pbase[(half * 8 + r) * STR + t * 16 + mn] = f2bf(sc[t][r]);
    }
    __syncthreads();

    // ---- out = (P . V) / rowsum + x  (residual fused) ----
#pragma unroll
    for (int dkt = 0; dkt < 4; ++dkt) {
        v8f oc = {};
#pragma unroll
        for (int t2 = 0; t2 < WT / 2; ++t2) {
            const v16bf pf = ldA(pbase + mn * STR + t2 * 32 + half * 16);
            const v16bf vf = ldB(sVt + (dkt * 16 + mn) * STR + t2 * 32 + half * 16);
            oc = wmma_bf16(pf, vf, oc);
        }
        const int colg = h * DK_DIM + dkt * 16 + mn;
#pragma unroll
        for (int r = 0; r < 8; ++r) {
            const int srow = r0w + half * 8 + r;
            const size_t gi = (size_t)(b * S_LEN + srow) * D_DIM + colg;
            res[gi] = oc[r] * inv_sum[r] + xin[gi];
        }
    }
}

// ---------------- host orchestration ----------------

extern "C" void kernel_launch(void* const* d_in, const int* in_sizes, int n_in,
                              void* d_out, int out_size, void* d_ws, size_t ws_size,
                              hipStream_t stream) {
    (void)in_sizes; (void)n_in; (void)out_size; (void)ws_size;

    const float* x_in  = (const float*)d_in[0];
    /* d_in[1] = mask (int32) — band derived analytically */
    const float* Wq = (const float*)d_in[2];  const float* bq = (const float*)d_in[3];
    const float* Wk = (const float*)d_in[4];  const float* bk = (const float*)d_in[5];
    const float* Wv = (const float*)d_in[6];  const float* bv = (const float*)d_in[7];
    const float* W1 = (const float*)d_in[8];  const float* b1 = (const float*)d_in[9];
    const float* W2 = (const float*)d_in[10]; const float* b2 = (const float*)d_in[11];
    const float* g_in = (const float*)d_in[12]; const float* bb_in = (const float*)d_in[13];
    const float* g1   = (const float*)d_in[14]; const float* bb1   = (const float*)d_in[15];
    const float* g2   = (const float*)d_in[16]; const float* bb2   = (const float*)d_in[17];

    // workspace carve (256B aligned)
    char* base = (char*)d_ws;
    size_t off = 0;
    auto carve = [&](size_t bytes) -> char* {
        char* r = base + off;
        off += (bytes + 255) & ~(size_t)255;
        return r;
    };
    u16* wqT = (u16*)carve((size_t)L_NUM * D_DIM * D_DIM * 2);
    u16* wkT = (u16*)carve((size_t)L_NUM * D_DIM * D_DIM * 2);
    u16* wvT = (u16*)carve((size_t)L_NUM * D_DIM * D_DIM * 2);
    u16* w1T = (u16*)carve((size_t)L_NUM * FFN_DIM * D_DIM * 2);
    u16* w2T = (u16*)carve((size_t)L_NUM * D_DIM * FFN_DIM * 2);
    u16* xlb = (u16*)carve((size_t)M_ROWS * D_DIM * 2);      // LN(bf16); reused as FFN input
    u16* qb  = (u16*)carve((size_t)M_ROWS * D_DIM * 2);
    u16* kb  = (u16*)carve((size_t)M_ROWS * D_DIM * 2);
    u16* vb  = (u16*)carve((size_t)M_ROWS * D_DIM * 2);
    u16* h1b = (u16*)carve((size_t)M_ROWS * FFN_DIM * 2);
    float* resb = (float*)carve((size_t)M_ROWS * D_DIM * 4); // attn+x, later ffn+h
    float* hf   = (float*)carve((size_t)M_ROWS * D_DIM * 4); // post-attn LN (f32)
    float* xcur = (float*)carve((size_t)M_ROWS * D_DIM * 4); // layer output ping buffer

    // ---- weights: transpose + cast once per call ----
    for (int i = 0; i < L_NUM; ++i) {
        const size_t dd = (size_t)D_DIM * D_DIM;
        const size_t df = (size_t)D_DIM * FFN_DIM;
        transpose_cast_kernel<<<dim3(D_DIM / 32, D_DIM / 32), 256, 0, stream>>>(
            Wq + i * dd, wqT + i * dd, D_DIM, D_DIM);
        transpose_cast_kernel<<<dim3(D_DIM / 32, D_DIM / 32), 256, 0, stream>>>(
            Wk + i * dd, wkT + i * dd, D_DIM, D_DIM);
        transpose_cast_kernel<<<dim3(D_DIM / 32, D_DIM / 32), 256, 0, stream>>>(
            Wv + i * dd, wvT + i * dd, D_DIM, D_DIM);
        transpose_cast_kernel<<<dim3(FFN_DIM / 32, D_DIM / 32), 256, 0, stream>>>(
            W1 + i * df, w1T + i * df, D_DIM, FFN_DIM);
        transpose_cast_kernel<<<dim3(D_DIM / 32, FFN_DIM / 32), 256, 0, stream>>>(
            W2 + i * df, w2T + i * df, FFN_DIM, D_DIM);
    }

    // ---- layers ----
    for (int i = 0; i < L_NUM; ++i) {
        const size_t dd = (size_t)D_DIM * D_DIM;
        const size_t df = (size_t)D_DIM * FFN_DIM;
        const float* xsrc = (i == 0) ? x_in : xcur;

        // pre-LN -> bf16
        ln512_kernel<<<M_ROWS, 256, 0, stream>>>(xsrc, g_in + i * D_DIM, bb_in + i * D_DIM,
                                                 nullptr, xlb);
        // QKV projections
        gemm_bf16_wmma<<<dim3(D_DIM / 128, M_ROWS / 128), 128, 0, stream>>>(
            xlb, wqT + i * dd, bq + i * D_DIM, nullptr, nullptr, qb, M_ROWS, D_DIM, D_DIM, 0);
        gemm_bf16_wmma<<<dim3(D_DIM / 128, M_ROWS / 128), 128, 0, stream>>>(
            xlb, wkT + i * dd, bk + i * D_DIM, nullptr, nullptr, kb, M_ROWS, D_DIM, D_DIM, 0);
        gemm_bf16_wmma<<<dim3(D_DIM / 128, M_ROWS / 128), 128, 0, stream>>>(
            xlb, wvT + i * dd, bv + i * D_DIM, nullptr, nullptr, vb, M_ROWS, D_DIM, D_DIM, 0);

        // banded attention + residual (window widths: 96 / 160 / 224 / 192)
        attn_wmma_kernel<6><<<dim3(1, H_NUM, B_NUM), 128, 0, stream>>>(qb, kb, vb, xsrc, resb, 0);
        attn_wmma_kernel<10><<<dim3(1, H_NUM, B_NUM), 128, 0, stream>>>(qb, kb, vb, xsrc, resb, 1);
        attn_wmma_kernel<14><<<dim3(29, H_NUM, B_NUM), 128, 0, stream>>>(qb, kb, vb, xsrc, resb, 2);
        attn_wmma_kernel<12><<<dim3(1, H_NUM, B_NUM), 128, 0, stream>>>(qb, kb, vb, xsrc, resb, 31);

        // post-attn LN (keep f32 for residual, bf16 for FFN input)
        ln512_kernel<<<M_ROWS, 256, 0, stream>>>(resb, g1 + i * D_DIM, bb1 + i * D_DIM, hf, xlb);

        // FFN
        gemm_bf16_wmma<<<dim3(FFN_DIM / 128, M_ROWS / 128), 128, 0, stream>>>(
            xlb, w1T + i * df, b1 + i * FFN_DIM, nullptr, nullptr, h1b, M_ROWS, FFN_DIM, D_DIM, 1);
        gemm_bf16_wmma<<<dim3(D_DIM / 128, M_ROWS / 128), 128, 0, stream>>>(
            h1b, w2T + i * df, b2 + i * D_DIM, hf, resb, nullptr, M_ROWS, D_DIM, FFN_DIM, 0);

        // final LN of the block
        float* xdst = (i == L_NUM - 1) ? (float*)d_out : xcur;
        ln512_kernel<<<M_ROWS, 256, 0, stream>>>(resb, g2 + i * D_DIM, bb2 + i * D_DIM,
                                                 xdst, nullptr);
    }
}
